// Gramba_29772713296337
// MI455X (gfx1250) — matrix operand order repaired and checked by reading, not compiled, not run on hardware
//
#include <hip/hip_runtime.h>
#include <hip/hip_bf16.h>
#include <stdint.h>

// ---------------------------------------------------------------------------
// Gramba block (minGRU MLP) for MI455X / gfx1250.
// GEMMs in bf16 via v_wmma_f32_16x16x32_bf16 (fp32 accumulate), staged into
// LDS with GLOBAL_LOAD_ASYNC_TO_LDS_B128 (ASYNCcnt-tracked, double buffered),
// chunked parallel scan for the minGRU recurrence.
// ---------------------------------------------------------------------------

typedef __attribute__((ext_vector_type(16))) __bf16 v16bf;
typedef __attribute__((ext_vector_type(8)))  float  v8f;

union FragBf {
    v16bf v;
    uint4 q[2];
};
static_assert(sizeof(FragBf) == 32, "frag size");

#define HID   512
#define EXPD  2048
#define MTOK  16384   // B*S = 4*4096
#define SEQ   4096
#define NBAT  4
#define CHUNK 128
#define NCHK  (SEQ / CHUNK)   // 32

#define GLOBAL_AS __attribute__((address_space(1)))
#define LDS_AS    __attribute__((address_space(3)))

// ---- CDNA5 async global->LDS copy (16B per lane), tracked by ASYNCcnt ----
__device__ __forceinline__ void async_copy_b128(void* lds, const void* g) {
    // vdst = 32-bit LDS byte offset, vaddr = 64-bit global address
    asm volatile("global_load_async_to_lds_b128 %0, %1, off"
                 :: "v"((unsigned)(uintptr_t)lds), "v"(g)
                 : "memory");
}

template <int N>
__device__ __forceinline__ void wait_asynccnt() {
#if __has_builtin(__builtin_amdgcn_s_wait_asynccnt)
    __builtin_amdgcn_s_wait_asynccnt(N);
#else
    asm volatile("s_wait_asynccnt %0" :: "i"(N) : "memory");
#endif
}

// ---------------------------------------------------------------------------
// fp32 -> bf16 straight convert
// ---------------------------------------------------------------------------
__global__ void cvt_bf16_kernel(const float* __restrict__ in,
                                __bf16* __restrict__ out, int n) {
    int i = blockIdx.x * blockDim.x + threadIdx.x;
    if (i < n) out[i] = (__bf16)in[i];
}

// fp32 (R x C) -> bf16 transposed (C x R)
__global__ void cvt_t_bf16_kernel(const float* __restrict__ in,
                                  __bf16* __restrict__ out, int R, int C) {
    int i = blockIdx.x * blockDim.x + threadIdx.x;
    if (i < R * C) {
        int c = i / R;       // output row (original column)
        int r = i % R;       // output col (original row)
        out[(size_t)c * R + r] = (__bf16)in[(size_t)r * C + c];
    }
}

// ---------------------------------------------------------------------------
// Tiled WMMA GEMM: D[M,N] = A[M,K] * Bt[N,K]^T (+ bias, + epilogue)
//   Block tile: 128(M) x 64(N), K-step 32, 256 threads = 8 waves (wave32).
//   Wave tile : 32x32 = 2x2 of v_wmma_f32_16x16x32_bf16.
//   Staging   : async global->LDS DMA, 2-deep software pipeline.
// MODE 0: dual-B, silu epilogue. out1 = bf16 x_in, out2 = fp32 x_skip.
// MODE 1: dual-B, minGRU gate epilogue. out1 = fp32 a = 1-sigmoid(zl),
//         out2 = fp32 b = sigmoid(zl)*hl.
// MODE 2: single-B, bias-only. out1 = fp32 result.
// ---------------------------------------------------------------------------
#define SA 40   // LDS row stride in halfs (32 data + 8 pad) => 80B, 16B aligned

template <int MODE>
__global__ __launch_bounds__(256)
void gemm_wmma_kernel(const __bf16* __restrict__ Ag,
                      const __bf16* __restrict__ B1g,
                      const __bf16* __restrict__ B2g,
                      const float*  __restrict__ bias1,
                      const float*  __restrict__ bias2,
                      void* __restrict__ out1v,
                      void* __restrict__ out2v,
                      int M, int N, int K) {
    __shared__ __align__(16) __bf16 As[2][128 * SA];
    __shared__ __align__(16) __bf16 Bs[2][2][64 * SA];

    const int tid  = threadIdx.x;
    const int lane = tid & 31;
    const int wave = tid >> 5;
    const int wm   = wave & 3;   // 0..3  (M)
    const int wn   = wave >> 2;  // 0..1  (N)

    const int m0 = blockIdx.y * 128;
    const int n0 = blockIdx.x * 64;

    // async ops issued per thread per tile (in-order completion on ASYNCcnt)
    constexpr int NIN = (MODE == 2) ? 3 : 4;

    // staging coordinates (16B chunks: row = idx/4, seg = idx%4)
    const int brow = tid >> 2;
    const int bseg = tid & 3;

    auto fill = [&](int buf, int k0) {
        #pragma unroll
        for (int it = 0; it < 2; ++it) {          // A: 512 chunks / 256 thr
            int idx = tid + it * 256;
            int row = idx >> 2;
            int seg = idx & 3;
            async_copy_b128(&As[buf][row * SA + seg * 8],
                            Ag + (size_t)(m0 + row) * K + k0 + seg * 8);
        }
        async_copy_b128(&Bs[buf][0][brow * SA + bseg * 8],
                        B1g + (size_t)(n0 + brow) * K + k0 + bseg * 8);
        if (MODE != 2)
            async_copy_b128(&Bs[buf][1][brow * SA + bseg * 8],
                            B2g + (size_t)(n0 + brow) * K + k0 + bseg * 8);
    };

    v8f c1[2][2] = {};
    v8f c2[2][2] = {};

    const int nk = K >> 5;
    fill(0, 0);   // prime the pipeline

    for (int t = 0; t < nk; ++t) {
        const int buf = t & 1;
        if (t + 1 < nk) {
            fill(buf ^ 1, (t + 1) << 5);   // issue next tile into other buffer
            wait_asynccnt<NIN>();          // retire tile t (in-order)
        } else {
            wait_asynccnt<0>();
        }
        __syncthreads();                    // publish tile t to all waves

        // ---- load fragments (ISA 7.12.2 16-bit layouts) ----
        // A: lane r<16 -> row r, K {0..7,16..23}; lane>=16 -> row r-16, K {8..15,24..31}
        FragBf aF[2];
        const int ar  = lane & 15;
        const int akb = (lane < 16) ? 0 : 8;
        #pragma unroll
        for (int mt = 0; mt < 2; ++mt) {
            int rbase = (wm * 32 + mt * 16 + ar) * SA;
            aF[mt].q[0] = *(const uint4*)&As[buf][rbase + akb];
            aF[mt].q[1] = *(const uint4*)&As[buf][rbase + akb + 16];
        }
        // B: lane n<16 -> col n, K {0..7,16..23}; lane>=16 -> col n-16, K {8..15,24..31}
        FragBf bF1[2], bF2[2];
        const int bkb = (lane < 16) ? 0 : 8;
        #pragma unroll
        for (int nt = 0; nt < 2; ++nt) {
            int rbase = (wn * 32 + nt * 16 + ar) * SA;
            bF1[nt].q[0] = *(const uint4*)&Bs[buf][0][rbase + bkb];
            bF1[nt].q[1] = *(const uint4*)&Bs[buf][0][rbase + bkb + 16];
            if (MODE != 2) {
                bF2[nt].q[0] = *(const uint4*)&Bs[buf][1][rbase + bkb];
                bF2[nt].q[1] = *(const uint4*)&Bs[buf][1][rbase + bkb + 16];
            }
        }
        // ---- WMMA ----
        #pragma unroll
        for (int mt = 0; mt < 2; ++mt) {
            #pragma unroll
            for (int nt = 0; nt < 2; ++nt) {
                c1[mt][nt] = __builtin_amdgcn_wmma_f32_16x16x32_bf16(
                    false, aF[mt].v, false, bF1[nt].v, (short)0, c1[mt][nt],
                    false, false);
                if (MODE != 2) {
                    c2[mt][nt] = __builtin_amdgcn_wmma_f32_16x16x32_bf16(
                        false, aF[mt].v, false, bF2[nt].v, (short)0, c2[mt][nt],
                        false, false);
                }
            }
        }
        __syncthreads();   // reads done before this buffer is refilled at t+1
    }

    // ---- epilogue ----
    // C/D layout: lane L, VGPR i -> row = 8*(L>=16) + i, col = L%16
    const int colLane = lane & 15;
    const int rowSel  = (lane >> 4) * 8;
    #pragma unroll
    for (int mt = 0; mt < 2; ++mt) {
        #pragma unroll
        for (int nt = 0; nt < 2; ++nt) {
            #pragma unroll
            for (int i = 0; i < 8; ++i) {
                size_t grow = (size_t)(m0 + wm * 32 + mt * 16 + rowSel + i);
                int    gcol = n0 + wn * 32 + nt * 16 + colLane;
                size_t o    = grow * (size_t)N + gcol;
                float v1 = c1[mt][nt][i] + bias1[gcol];
                if (MODE == 0) {
                    float v2 = c2[mt][nt][i] + bias2[gcol];
                    float s1 = v1 / (1.f + __expf(-v1));   // silu
                    float s2 = v2 / (1.f + __expf(-v2));
                    ((__bf16*)out1v)[o] = (__bf16)s1;      // x_in (bf16 GEMM input)
                    ((float*)out2v)[o]  = s2;              // x_skip (fp32)
                } else if (MODE == 1) {
                    float hl = c2[mt][nt][i] + bias2[gcol];
                    float z  = 1.f / (1.f + __expf(-v1));  // sigmoid(zl)
                    ((float*)out1v)[o] = 1.f - z;          // a coefficient
                    ((float*)out2v)[o] = z * hl;           // b term
                } else {
                    ((float*)out1v)[o] = v1;               // final output
                }
            }
        }
    }
}

// ---------------------------------------------------------------------------
// minGRU scan: h_t = a_t*h_{t-1} + b_t, chunked 3-pass parallel prefix.
// Pass 1: per-chunk local scan; overwrite a with running prefix-product,
//         b with local h (zero carry-in); emit per-chunk carries.
// ---------------------------------------------------------------------------
__global__ void scan_chunk_kernel(float* __restrict__ a, float* __restrict__ b,
                                  float* __restrict__ Ac, float* __restrict__ Bc) {
    const int tid = threadIdx.x;
    const int chk = blockIdx.x;        // 0..31
    const int fg  = blockIdx.y;        // 0..7
    const int bb  = blockIdx.z;        // 0..3
    const int f   = fg * 256 + tid;
    size_t base = ((size_t)bb * SEQ + (size_t)chk * CHUNK) * EXPD + f;
    float ap = 1.f, h = 0.f;
    for (int s = 0; s < CHUNK; ++s) {
        size_t o = base + (size_t)s * EXPD;
        float av = a[o], bv = b[o];
        h  = __builtin_fmaf(av, h, bv);
        ap = ap * av;
        a[o] = ap;
        b[o] = h;
    }
    size_t ci = ((size_t)bb * NCHK + chk) * EXPD + f;
    Ac[ci] = ap;
    Bc[ci] = h;
}

// Pass 2: serial scan over the 32 chunk carries -> per-chunk carry-in state.
__global__ void scan_carry_kernel(const float* __restrict__ Ac,
                                  const float* __restrict__ Bc,
                                  float* __restrict__ Hin) {
    const int f  = blockIdx.x * 256 + threadIdx.x;   // 0..2047
    const int bb = blockIdx.y;                       // 0..3
    float h = 0.f;
    for (int c = 0; c < NCHK; ++c) {
        size_t ci = ((size_t)bb * NCHK + c) * EXPD + f;
        Hin[ci] = h;
        h = __builtin_fmaf(Ac[ci], h, Bc[ci]);
    }
}

// Pass 3: h = h_local + aprefix*carry_in; fuse s = x_skip + h -> bf16.
__global__ void scan_apply_kernel(const float* __restrict__ a,
                                  const float* __restrict__ b,
                                  const float* __restrict__ Hin,
                                  const float* __restrict__ xskip,
                                  __bf16* __restrict__ sb) {
    const int tid = threadIdx.x;
    const int chk = blockIdx.x;
    const int fg  = blockIdx.y;
    const int bb  = blockIdx.z;
    const int f   = fg * 256 + tid;
    const float hin = Hin[((size_t)bb * NCHK + chk) * EXPD + f];
    size_t base = ((size_t)bb * SEQ + (size_t)chk * CHUNK) * EXPD + f;
    for (int s = 0; s < CHUNK; ++s) {
        size_t o = base + (size_t)s * EXPD;
        float h = __builtin_fmaf(a[o], hin, b[o]);
        sb[o] = (__bf16)(xskip[o] + h);
    }
}

// ---------------------------------------------------------------------------
// launch
// ---------------------------------------------------------------------------
extern "C" void kernel_launch(void* const* d_in, const int* in_sizes, int n_in,
                              void* d_out, int out_size, void* d_ws, size_t ws_size,
                              hipStream_t stream) {
    const float* x     = (const float*)d_in[0];
    const float* W_in1 = (const float*)d_in[1];
    const float* b_in1 = (const float*)d_in[2];
    const float* W_in2 = (const float*)d_in[3];
    const float* b_in2 = (const float*)d_in[4];
    const float* W_z   = (const float*)d_in[5];
    const float* b_z   = (const float*)d_in[6];
    const float* W_h   = (const float*)d_in[7];
    const float* b_h   = (const float*)d_in[8];
    const float* W_out = (const float*)d_in[9];
    const float* b_out = (const float*)d_in[10];
    float* out = (float*)d_out;

    // workspace carve-up (256B aligned)
    char* w = (char*)d_ws;
    size_t off = 0;
    auto alloc = [&](size_t bytes) {
        void* p = w + off;
        off = (off + bytes + 255) & ~(size_t)255;
        return p;
    };
    __bf16* xb   = (__bf16*)alloc((size_t)MTOK * HID * 2);
    __bf16* W1t  = (__bf16*)alloc((size_t)EXPD * HID * 2);
    __bf16* W2t  = (__bf16*)alloc((size_t)EXPD * HID * 2);
    __bf16* Wzt  = (__bf16*)alloc((size_t)EXPD * EXPD * 2);
    __bf16* Wht  = (__bf16*)alloc((size_t)EXPD * EXPD * 2);
    __bf16* Wot  = (__bf16*)alloc((size_t)HID * EXPD * 2);
    __bf16* xin  = (__bf16*)alloc((size_t)MTOK * EXPD * 2);
    float*  xskp = (float*) alloc((size_t)MTOK * EXPD * 4);
    float*  aC   = (float*) alloc((size_t)MTOK * EXPD * 4);
    float*  bC   = (float*) alloc((size_t)MTOK * EXPD * 4);
    __bf16* sb   = (__bf16*)alloc((size_t)MTOK * EXPD * 2);
    float*  Ac   = (float*) alloc((size_t)NBAT * NCHK * EXPD * 4);
    float*  Bc   = (float*) alloc((size_t)NBAT * NCHK * EXPD * 4);
    float*  Hin  = (float*) alloc((size_t)NBAT * NCHK * EXPD * 4);
    (void)ws_size; (void)n_in; (void)in_sizes; (void)out_size;

    // 1) bf16 conversions / transposes
    {
        int n = MTOK * HID;
        cvt_bf16_kernel<<<(n + 255) / 256, 256, 0, stream>>>(x, xb, n);
    }
    cvt_t_bf16_kernel<<<(HID * EXPD + 255) / 256, 256, 0, stream>>>(W_in1, W1t, HID, EXPD);
    cvt_t_bf16_kernel<<<(HID * EXPD + 255) / 256, 256, 0, stream>>>(W_in2, W2t, HID, EXPD);
    cvt_t_bf16_kernel<<<(EXPD * EXPD + 255) / 256, 256, 0, stream>>>(W_z, Wzt, EXPD, EXPD);
    cvt_t_bf16_kernel<<<(EXPD * EXPD + 255) / 256, 256, 0, stream>>>(W_h, Wht, EXPD, EXPD);
    cvt_t_bf16_kernel<<<(EXPD * HID + 255) / 256, 256, 0, stream>>>(W_out, Wot, EXPD, HID);

    // 2) x_in = silu(x@W1+b1) [bf16], x_skip = silu(x@W2+b2) [f32]
    gemm_wmma_kernel<0><<<dim3(EXPD / 64, MTOK / 128), 256, 0, stream>>>(
        xb, W1t, W2t, b_in1, b_in2, (void*)xin, (void*)xskp, MTOK, EXPD, HID);

    // 3) a = 1-sigmoid(x_in@Wz+bz), b = sigmoid(.)*(x_in@Wh+bh)
    gemm_wmma_kernel<1><<<dim3(EXPD / 64, MTOK / 128), 256, 0, stream>>>(
        xin, Wzt, Wht, b_z, b_h, (void*)aC, (void*)bC, MTOK, EXPD, EXPD);

    // 4) chunked parallel scan + fuse s = x_skip + h (bf16)
    scan_chunk_kernel<<<dim3(NCHK, EXPD / 256, NBAT), 256, 0, stream>>>(aC, bC, Ac, Bc);
    scan_carry_kernel<<<dim3(EXPD / 256, NBAT), 256, 0, stream>>>(Ac, Bc, Hin);
    scan_apply_kernel<<<dim3(NCHK, EXPD / 256, NBAT), 256, 0, stream>>>(aC, bC, Hin, xskp, sb);

    // 5) out = s @ W_out + b_out
    gemm_wmma_kernel<2><<<dim3(HID / 64, MTOK / 128), 256, 0, stream>>>(
        sb, Wot, nullptr, b_out, nullptr, (void*)out, nullptr, MTOK, HID, EXPD);
}
